// SwinTransformer_MAE3D_57904749085153
// MI455X (gfx1250) — matrix operand order repaired and checked by reading, not compiled, not run on hardware
//
#include <hip/hip_runtime.h>
#include <hip/hip_bf16.h>

typedef __attribute__((ext_vector_type(16))) _Float16 v16h;
typedef __attribute__((ext_vector_type(8)))  _Float16 v8h;
typedef __attribute__((ext_vector_type(8)))  float    v8f;

#define NWIN  64
#define NTOK  512
#define CCH   256
#define NHEAD 8
#define HDIM  32

static __device__ __forceinline__ v8f wmma_f16(v16h a, v16h b, v8f c) {
    // D = A(16x32 f16) x B(32x16 f16) + C(16x16 f32)
    return __builtin_amdgcn_wmma_f32_16x16x32_f16(false, a, false, b, (short)0, c, false, false);
}

static __device__ __forceinline__ int region1(int g) {
    // shifted-window mask region along one axis (H=32, WS=8, SS=4)
    return (g < 24) ? 0 : ((g < 28) ? 1 : 2);
}

// ---------------------------------------------------------------- prep ----
__global__ void k_f32_to_f16(const float* __restrict__ src, _Float16* __restrict__ dst, int n) {
    int i = blockIdx.x * blockDim.x + threadIdx.x;
    if (i < n) dst[i] = (_Float16)src[i];
}

// cyclic shift (-4,-4,-4) + window partition -> xw [64*512, 256] f16
__global__ void k_gather_shift(const float* __restrict__ x, _Float16* __restrict__ xw) {
    int i = blockIdx.x * blockDim.x + threadIdx.x;
    int c = i & 255;
    int t = (i >> 8) & 511;
    int w = i >> 17;
    int wd = w & 3, ww = (w >> 2) & 3, wh = w >> 4;
    int a = t >> 6, b = (t >> 3) & 7, cc = t & 7;
    int gh = (wh * 8 + a + 4) & 31;
    int gw = (ww * 8 + b + 4) & 31;
    int gd = (wd * 8 + cc + 4) & 31;
    xw[i] = (_Float16)x[((gh * 32 + gw) * 32 + gd) * 256 + c];
}

// ------------------------------------------------------------ QKV GEMM ----
// C[32768,768] = A[32768,256] * W[768,256]^T + bias ; scatter into q/k/vT
__global__ void k_gemm_qkv(const _Float16* __restrict__ A, const _Float16* __restrict__ W,
                           const float* __restrict__ bias,
                           _Float16* __restrict__ qh, _Float16* __restrict__ kh,
                           _Float16* __restrict__ vTh) {
    const int NT_N = 768 / 16;
    int wave = (blockIdx.x * blockDim.x + threadIdx.x) >> 5;
    int lane = threadIdx.x & 31;
    int hi = lane >> 4, lo = lane & 15;
    int tm = wave / NT_N;
    int tn = wave - tm * NT_N;
    int arow = tm * 16 + lo;     // A-frag: M = lane&15
    int bcol = tn * 16 + lo;     // B-frag: N = lane&15

    v8f acc = {};
#pragma unroll
    for (int kk = 0; kk < 256; kk += 32) {
        v8h a0 = *(const v8h*)(A + (size_t)arow * 256 + kk + hi * 16);
        v8h a1 = *(const v8h*)(A + (size_t)arow * 256 + kk + hi * 16 + 8);
        v16h af;
#pragma unroll
        for (int t = 0; t < 8; ++t) { af[t] = a0[t]; af[t + 8] = a1[t]; }
        v16h bf = *(const v16h*)(W + (size_t)bcol * 256 + kk + hi * 16);
        acc = wmma_f16(af, bf, acc);
    }

    int j = bcol;                 // output column 0..767
    int part = j >> 8;            // 0=q 1=k 2=v
    int r = j & 255;
    int head = r >> 5;
    int d = r & 31;
    float bj = bias[j];
#pragma unroll
    for (int e = 0; e < 8; ++e) {
        int row = tm * 16 + hi * 8 + e;   // C-layout: M = e + 8*hi
        int w = row >> 9;
        int tok = row & 511;
        float v = acc[e] + bj;
        size_t wh_base = (size_t)(w * NHEAD + head);
        if (part == 0) {
            v *= 0.17677669529663687f;    // 32^-0.5
            qh[(wh_base * 512 + tok) * 32 + d] = (_Float16)v;
        } else if (part == 1) {
            kh[(wh_base * 512 + tok) * 32 + d] = (_Float16)v;
        } else {
            vTh[(wh_base * 32 + d) * 512 + tok] = (_Float16)v;  // transposed
        }
    }
}

// ----------------------------------------------------------- attention ----
// one wave = one 16-query strip of one (window, head); flash softmax over
// 16 key-pairs of 32; bias+mask computed on the fly.
__global__ void k_attn(const _Float16* __restrict__ qh, const _Float16* __restrict__ kh,
                       const _Float16* __restrict__ vTh, const float* __restrict__ rpb,
                       _Float16* __restrict__ ao) {
    __shared__ _Float16 pstage[8][16 * 32];   // 1KB per wave P staging
    int wv = threadIdx.x >> 5;
    int lane = threadIdx.x & 31;
    int hi = lane >> 4, lo = lane & 15;

    int strip = blockIdx.x * 8 + wv;          // ((w*8+h)*32 + sb)
    int w = strip >> 8;
    int h = (strip >> 5) & 7;
    int sbase = (strip & 31) << 4;

    const _Float16* qp = qh + (size_t)(w * NHEAD + h) * 512 * 32;
    const _Float16* kp = kh + (size_t)(w * NHEAD + h) * 512 * 32;
    const _Float16* vp = vTh + (size_t)(w * NHEAD + h) * 32 * 512;

    int wwd = w & 3, www = (w >> 2) & 3, wwh = w >> 4;

    // q fragment (A layout, M = lane&15, two 8-half segments)
    int qrow = sbase + lo;
    v16h qa;
    {
        v8h a0 = *(const v8h*)(qp + (size_t)qrow * 32 + hi * 16);
        v8h a1 = *(const v8h*)(qp + (size_t)qrow * 32 + hi * 16 + 8);
#pragma unroll
        for (int t = 0; t < 8; ++t) { qa[t] = a0[t]; qa[t + 8] = a1[t]; }
    }

    // per-row query token info (C-layout rows: e + 8*hi)
    int an[8], bn[8], cn[8], cntn[8];
#pragma unroll
    for (int e = 0; e < 8; ++e) {
        int tn = sbase + hi * 8 + e;
        an[e] = tn >> 6; bn[e] = (tn >> 3) & 7; cn[e] = tn & 7;
        cntn[e] = region1(wwh * 8 + an[e]) * 9 + region1(www * 8 + bn[e]) * 3 +
                  region1(wwd * 8 + cn[e]);
    }

    float mrow[8], lrow[8];
    v8f o0 = {}, o1 = {};
#pragma unroll
    for (int e = 0; e < 8; ++e) { mrow[e] = -1e30f; lrow[e] = 0.f; }

    _Float16* ps = &pstage[wv][0];

    for (int p = 0; p < 16; ++p) {
        int kb = p * 32;
        float snew[2][8];
#pragma unroll
        for (int s = 0; s < 2; ++s) {
            int key = kb + s * 16 + lo;                       // B-frag: N = lane&15
            v16h bfk = *(const v16h*)(kp + (size_t)key * 32 + hi * 16);
            v8f sc = {};
            sc = wmma_f16(qa, bfk, sc);                       // S = q * k^T
            int am = key >> 6, bm = (key >> 3) & 7, cm = key & 7;
            int cntm = region1(wwh * 8 + am) * 9 + region1(www * 8 + bm) * 3 +
                       region1(wwd * 8 + cm);
#pragma unroll
            for (int e = 0; e < 8; ++e) {
                int idx = ((an[e] - am + 7) * 15 + (bn[e] - bm + 7)) * 15 + (cn[e] - cm + 7);
                float bv = rpb[idx * NHEAD + h];
                float msk = (cntm == cntn[e]) ? 0.f : -100.f;
                snew[s][e] = sc[e] + bv + msk;
            }
        }
        // online softmax update per row
        float pv[2][8];
#pragma unroll
        for (int e = 0; e < 8; ++e) {
            float tmax = fmaxf(snew[0][e], snew[1][e]);
#pragma unroll
            for (int off = 1; off < 16; off <<= 1)
                tmax = fmaxf(tmax, __shfl_xor(tmax, off, 32));
            float mnew = fmaxf(mrow[e], tmax);
            float alpha = __expf(mrow[e] - mnew);
            mrow[e] = mnew;
            float p0 = __expf(snew[0][e] - mnew);
            float p1 = __expf(snew[1][e] - mnew);
            float psum = p0 + p1;
#pragma unroll
            for (int off = 1; off < 16; off <<= 1)
                psum += __shfl_xor(psum, off, 32);
            lrow[e] = lrow[e] * alpha + psum;
            o0[e] *= alpha; o1[e] *= alpha;
            pv[0][e] = p0; pv[1][e] = p1;
        }
        // stage P (C layout -> LDS) then reload in A layout (wave-private region)
#pragma unroll
        for (int s = 0; s < 2; ++s)
#pragma unroll
            for (int e = 0; e < 8; ++e)
                ps[(hi * 8 + e) * 32 + s * 16 + lo] = (_Float16)pv[s][e];
        v16h pa;
        {
            v8h a0 = *(const v8h*)(ps + lo * 32 + hi * 16);
            v8h a1 = *(const v8h*)(ps + lo * 32 + hi * 16 + 8);
#pragma unroll
            for (int t = 0; t < 8; ++t) { pa[t] = a0[t]; pa[t + 8] = a1[t]; }
        }
        // O += P * V  (V transposed: contiguous 16-half B segments)
        v16h bv0 = *(const v16h*)(vp + (size_t)lo * 512 + kb + hi * 16);
        v16h bv1 = *(const v16h*)(vp + (size_t)(lo + 16) * 512 + kb + hi * 16);
        o0 = wmma_f16(pa, bv0, o0);
        o1 = wmma_f16(pa, bv1, o1);
    }

    // normalize and write head output [w, tok, 256] at column head*32 + d
#pragma unroll
    for (int e = 0; e < 8; ++e) {
        float inv = 1.f / lrow[e];
        int tok = sbase + hi * 8 + e;
        size_t base = ((size_t)w * 512 + tok) * 256 + h * 32;
        ao[base + lo] = (_Float16)(o0[e] * inv);
        ao[base + 16 + lo] = (_Float16)(o1[e] * inv);
    }
}

// ----------------------------------------------------------- proj GEMM ----
// out[32768,256] = AO[32768,256] * Wp[256,256]^T + bias ; un-window + unshift
__global__ void k_gemm_proj(const _Float16* __restrict__ A, const _Float16* __restrict__ W,
                            const float* __restrict__ bias, float* __restrict__ out) {
    const int NT_N = 256 / 16;
    int wave = (blockIdx.x * blockDim.x + threadIdx.x) >> 5;
    int lane = threadIdx.x & 31;
    int hi = lane >> 4, lo = lane & 15;
    int tm = wave >> 4;
    int tn = wave & (NT_N - 1);
    int arow = tm * 16 + lo;
    int bcol = tn * 16 + lo;

    v8f acc = {};
#pragma unroll
    for (int kk = 0; kk < 256; kk += 32) {
        v8h a0 = *(const v8h*)(A + (size_t)arow * 256 + kk + hi * 16);
        v8h a1 = *(const v8h*)(A + (size_t)arow * 256 + kk + hi * 16 + 8);
        v16h af;
#pragma unroll
        for (int t = 0; t < 8; ++t) { af[t] = a0[t]; af[t + 8] = a1[t]; }
        v16h bf = *(const v16h*)(W + (size_t)bcol * 256 + kk + hi * 16);
        acc = wmma_f16(af, bf, acc);
    }

    float bj = bias[bcol];
#pragma unroll
    for (int e = 0; e < 8; ++e) {
        int row = tm * 16 + hi * 8 + e;
        int w = row >> 9;
        int t = row & 511;
        int wd = w & 3, ww = (w >> 2) & 3, wh = w >> 4;
        int a = t >> 6, b = (t >> 3) & 7, cc = t & 7;
        int gh = (wh * 8 + a + 4) & 31;   // reverse cyclic shift (+4)
        int gw = (ww * 8 + b + 4) & 31;
        int gd = (wd * 8 + cc + 4) & 31;
        out[((size_t)(gh * 32 + gw) * 32 + gd) * 256 + bcol] = acc[e] + bj;
    }
}

// -------------------------------------------------------------- launch ----
extern "C" void kernel_launch(void* const* d_in, const int* in_sizes, int n_in,
                              void* d_out, int out_size, void* d_ws, size_t ws_size,
                              hipStream_t stream) {
    const float* x    = (const float*)d_in[0];
    const float* wqkv = (const float*)d_in[1];
    const float* bqkv = (const float*)d_in[2];
    const float* wprj = (const float*)d_in[3];
    const float* bprj = (const float*)d_in[4];
    const float* rpb  = (const float*)d_in[5];
    float* out = (float*)d_out;

    char* p = (char*)d_ws;
    auto alloc = [&](size_t bytes) {
        char* r = p;
        p += (bytes + 255) & ~(size_t)255;
        return r;
    };
    _Float16* xw_h   = (_Float16*)alloc((size_t)32768 * 256 * 2);
    _Float16* wqkv_h = (_Float16*)alloc((size_t)768 * 256 * 2);
    _Float16* wprj_h = (_Float16*)alloc((size_t)256 * 256 * 2);
    _Float16* q_h    = (_Float16*)alloc((size_t)NWIN * NHEAD * 512 * 32 * 2);
    _Float16* k_h    = (_Float16*)alloc((size_t)NWIN * NHEAD * 512 * 32 * 2);
    _Float16* vT_h   = (_Float16*)alloc((size_t)NWIN * NHEAD * 32 * 512 * 2);
    _Float16* ao_h   = (_Float16*)alloc((size_t)32768 * 256 * 2);

    k_f32_to_f16<<<(768 * 256) / 256, 256, 0, stream>>>(wqkv, wqkv_h, 768 * 256);
    k_f32_to_f16<<<(256 * 256) / 256, 256, 0, stream>>>(wprj, wprj_h, 256 * 256);
    k_gather_shift<<<(32768 * 256) / 256, 256, 0, stream>>>(x, xw_h);

    // 2048 x 48 tiles, 8 waves/block
    k_gemm_qkv<<<(2048 * 48) / 8, 256, 0, stream>>>(xw_h, wqkv_h, bqkv, q_h, k_h, vT_h);

    // 64 windows * 8 heads * 32 strips / 8 waves per block
    k_attn<<<(NWIN * NHEAD * 32) / 8, 256, 0, stream>>>(q_h, k_h, vT_h, rpb, ao_h);

    // 2048 x 16 tiles, 8 waves/block
    k_gemm_proj<<<(2048 * 16) / 8, 256, 0, stream>>>(ao_h, wprj_h, bprj, out);
}
